// MultibandFrameAttention_28578712387591
// MI455X (gfx1250) — matrix operand order repaired and checked by reading, not compiled, not run on hardware
//
#include <hip/hip_runtime.h>
#include <hip/hip_bf16.h>

// ---------------------------------------------------------------------------
// MultibandFrameAttention for MI455X (gfx1250), bf16 WMMA pipeline.
// B=1, W=2048, C=BINS=1024, H=16, d=64, K(conv)=3.
// ---------------------------------------------------------------------------

#define SEQ   2048
#define CH    1024
#define HEADS 16
#define DH    64

typedef __bf16 bf16;
typedef __attribute__((ext_vector_type(16))) __bf16 v16bf;
typedef __attribute__((ext_vector_type(8)))  __bf16 v8bf;
typedef __attribute__((ext_vector_type(4)))  __bf16 v4bf;
typedef __attribute__((ext_vector_type(8)))  float  v8f;
typedef __attribute__((ext_vector_type(4)))  int    v4i;

// ---- CDNA5 async global->LDS staging (ASYNCcnt-tracked) -------------------
#if defined(__gfx1250__) && __has_builtin(__builtin_amdgcn_global_load_async_to_lds_b128)
#define HAVE_ASYNC_LDS 1
#else
#define HAVE_ASYNC_LDS 0
#endif

#define AS_GLOBAL __attribute__((address_space(1)))
#define AS_LOCAL  __attribute__((address_space(3)))

// copy 16 bytes (8 x bf16) global -> LDS, per lane
__device__ static inline void cp16_g2l(const bf16* g, bf16* l) {
#if HAVE_ASYNC_LDS
  __builtin_amdgcn_global_load_async_to_lds_b128(
      (AS_GLOBAL v4i*)(uintptr_t)g,
      (AS_LOCAL  v4i*)(unsigned)(uintptr_t)l, 0, 0);
#else
  *(v8bf*)l = *(const v8bf*)g;
#endif
}

__device__ static inline void wait_async_stage() {
#if HAVE_ASYNC_LDS
#if __has_builtin(__builtin_amdgcn_s_wait_asynccnt)
  __builtin_amdgcn_s_wait_asynccnt(0);
#endif
#endif
}

// ---- WMMA fragment loaders (ISA 7.12.2 layouts, wave32) -------------------
// A matrix 16x32 bf16 from row-major LDS tile (ld = row stride in elems).
// lane L (0..15): row L, K=0..7 and 16..23; lane L+16: row L, K=8..15 and 24..31.
__device__ static inline v16bf load_a_frag(const bf16* base, int ld) {
  const int lane = threadIdx.x & 31;
  const int row  = lane & 15;
  const int half = lane >> 4;
  const bf16* p = base + row * ld + half * 8;
  v8bf lo = *(const v8bf*)(p);
  v8bf hi = *(const v8bf*)(p + 16);
  return __builtin_shufflevector(lo, hi, 0, 1, 2, 3, 4, 5, 6, 7,
                                 8, 9, 10, 11, 12, 13, 14, 15);
}

// B matrix 32x16, element B[k][n] stored K-contiguous: elem(n,k) = base[n*ld + k].
// lanes 0..15: col n=lane, K=0..7 and 16..23; lanes 16..31: col n=lane-16, K=8..15 and 24..31.
// Per-lane read is 2 x 8 contiguous bf16 -> 2 x ds_load_b128.
__device__ static inline v16bf load_b_kcontig(const bf16* base, int ld) {
  const int lane = threadIdx.x & 31;
  const bf16* p = base + (lane & 15) * ld + (lane >> 4) * 8;
  v8bf lo = *(const v8bf*)(p);
  v8bf hi = *(const v8bf*)(p + 16);
  return __builtin_shufflevector(lo, hi, 0, 1, 2, 3, 4, 5, 6, 7,
                                 8, 9, 10, 11, 12, 13, 14, 15);
}

#define WMMA_BF16(a, b, c) \
  __builtin_amdgcn_wmma_f32_16x16x32_bf16(false, (a), false, (b), (short)0, (c), false, false)

// ---------------------------------------------------------------------------
// f32 -> bf16 conversion, 4 elements/thread (n must be a multiple of 4)
// ---------------------------------------------------------------------------
__global__ void cvt_f32_bf16(const float* __restrict__ in, bf16* __restrict__ out, int n4) {
  for (int i = blockIdx.x * blockDim.x + threadIdx.x; i < n4; i += gridDim.x * blockDim.x) {
    float4 f = ((const float4*)in)[i];
    v4bf o;
    o[0] = (bf16)f.x; o[1] = (bf16)f.y; o[2] = (bf16)f.z; o[3] = (bf16)f.w;
    ((v4bf*)out)[i] = o;
  }
}

// er [DH][SEQ] f32 -> ERt [SEQ][DH] bf16
__global__ void transpose_er(const float* __restrict__ er, bf16* __restrict__ ert) {
  int i = blockIdx.x * blockDim.x + threadIdx.x;   // i over SEQ*DH
  if (i >= SEQ * DH) return;
  int w = i / DH, d = i % DH;
  ert[w * DH + d] = (bf16)er[d * SEQ + w];
}

// ---------------------------------------------------------------------------
// Tiled bf16 GEMM: C[M,N] (f32) = A[M,K] * B[K,N], row-major bf16 inputs.
// Block = 128 threads (4 waves). Tile 64x64. Wave w owns 16 rows x 64 cols.
// A tile staged row-major via async b128; B tile staged TRANSPOSED so
// B-fragment loads are K-contiguous.
// ---------------------------------------------------------------------------
__global__ __launch_bounds__(128) void gemm_bf16_nn(const bf16* __restrict__ A,
                                                    const bf16* __restrict__ Bm,
                                                    float* __restrict__ Cm,
                                                    int M, int N, int Kc) {
  __shared__ bf16 As[64 * 32];    // [row][k]
  __shared__ bf16 Bts[64 * 32];   // transposed: [n][k]
  const int m0 = blockIdx.y * 64;
  const int n0 = blockIdx.x * 64;
  const int tid = threadIdx.x, lane = tid & 31, wave = tid >> 5;
  const int wm = wave * 16;

  v8f acc[4] = {};
  for (int kk = 0; kk < Kc; kk += 32) {
    // A tile 64x32: 256 chunks of 8 bf16, contiguous both sides -> async b128
    for (int ch = tid; ch < 256; ch += 128) {
      int r = ch >> 2, c = (ch & 3) * 8;
      cp16_g2l(&A[(size_t)(m0 + r) * Kc + kk + c], &As[r * 32 + c]);
    }
    // B tile 32x64 -> transposed LDS [n][k]; coalesced global v8 reads
    for (int ch = tid; ch < 256; ch += 128) {
      int k = ch >> 3, n = (ch & 7) * 8;
      v8bf v = *(const v8bf*)&Bm[(size_t)(kk + k) * N + n0 + n];
#pragma unroll
      for (int j = 0; j < 8; ++j) Bts[(n + j) * 32 + k] = v[j];
    }
    if (kk + 32 < Kc)
      __builtin_prefetch(&A[(size_t)(m0 + (tid >> 1)) * Kc + kk + 32], 0, 1);
    wait_async_stage();
    __syncthreads();

    v16bf a = load_a_frag(As + wm * 32, 32);
#pragma unroll
    for (int t = 0; t < 4; ++t) {
      v16bf b = load_b_kcontig(Bts + (t * 16) * 32, 32);
      acc[t] = WMMA_BF16(a, b, acc[t]);
    }
    __syncthreads();
  }

  const int rbase = 8 * (lane >> 4);
  const int col   = lane & 15;
#pragma unroll
  for (int t = 0; t < 4; ++t)
#pragma unroll
    for (int r = 0; r < 8; ++r)
      Cm[(size_t)(m0 + wm + rbase + r) * N + n0 + t * 16 + col] = acc[t][r];
}

// ---------------------------------------------------------------------------
// Depthwise conv1d k=3, 'same' (zero pad), along W; 4 channels per thread.
// In f32 [SEQ][CH], kernel f32 [CH][1][3]. Output bf16.
// ---------------------------------------------------------------------------
__global__ void dwconv3_bf16(const float* __restrict__ in, const float* __restrict__ cw,
                             bf16* __restrict__ out) {
  int i = blockIdx.x * blockDim.x + threadIdx.x;   // over SEQ*CH/4
  if (i >= SEQ * CH / 4) return;
  int base = i * 4;
  int w = base / CH, c = base % CH;
  float4 mid = ((const float4*)in)[i];
  float4 lft = (w > 0)       ? ((const float4*)in)[i - CH / 4] : make_float4(0, 0, 0, 0);
  float4 rgt = (w < SEQ - 1) ? ((const float4*)in)[i + CH / 4] : make_float4(0, 0, 0, 0);
  v4bf o;
#pragma unroll
  for (int j = 0; j < 4; ++j) {
    const float* k3 = cw + (c + j) * 3;
    float l = (&lft.x)[j], m = (&mid.x)[j], r = (&rgt.x)[j];
    o[j] = (bf16)(k3[0] * l + k3[1] * m + k3[2] * r);
  }
  ((v4bf*)out)[i] = o;
}

// ---------------------------------------------------------------------------
// Flash-style attention per (head, 64-query block).
// S[i,j] = (q_i . k_j + ert_i . q_j) / sqrt(C); softmax over j; O = P.V
// Block = 128 threads (4 waves); each wave owns 16 query rows.
// ---------------------------------------------------------------------------
__global__ __launch_bounds__(128) void attn_flash(const bf16* __restrict__ Q,
                                                  const bf16* __restrict__ K,
                                                  const bf16* __restrict__ V,
                                                  const bf16* __restrict__ ERt,
                                                  float* __restrict__ O) {
  __shared__ bf16 Aq[64 * DH];     // query-block Q rows       [row][d]
  __shared__ bf16 Ae[64 * DH];     // query-block er^T rows    [row][d]
  __shared__ bf16 Kt[32 * DH];     // key-block K rows         [key][d]
  __shared__ bf16 Qk[32 * DH];     // key-block Q rows (skew)  [key][d]
  __shared__ bf16 Vtt[DH * 32];    // key-block V TRANSPOSED   [d][key]
  __shared__ float Sf[64 * 32];    // raw scores               [row][key]
  __shared__ bf16  Pt[64 * 32];    // probabilities            [row][key]
  __shared__ float mrow[64], lrow[64], arow[64];

  const int qb = blockIdx.x;
  const int h  = blockIdx.y;
  const int i0 = qb * 64;
  const int tid = threadIdx.x, lane = tid & 31, wave = tid >> 5;
  const int wm = wave * 16;
  const size_t hoff = (size_t)h * DH;
  const float scale = 0.03125f;    // 1/sqrt(1024)

  // stage query-block tiles (contiguous 16B chunks -> async b128)
  for (int ch = tid; ch < 512; ch += 128) {
    int r = ch >> 3, c = (ch & 7) * 8;
    cp16_g2l(&Q[(size_t)(i0 + r) * CH + hoff + c], &Aq[r * DH + c]);
    cp16_g2l(&ERt[(size_t)(i0 + r) * DH + c], &Ae[r * DH + c]);
  }
  if (tid < 64) { mrow[tid] = -1e30f; lrow[tid] = 0.f; }

  v8f acc[4] = {};
  const int rbase = 8 * (lane >> 4);
  const int col   = lane & 15;

  for (int j0 = 0; j0 < SEQ; j0 += 32) {
    __syncthreads();   // previous iteration's consumers done before restaging
    for (int ch = tid; ch < 256; ch += 128) {
      int r = ch >> 3, c = (ch & 7) * 8;
      cp16_g2l(&K[(size_t)(j0 + r) * CH + hoff + c], &Kt[r * DH + c]);
      cp16_g2l(&Q[(size_t)(j0 + r) * CH + hoff + c], &Qk[r * DH + c]);
      // V transposed into [d][key] so P.V B-fragments are K-contiguous
      v8bf v = *(const v8bf*)&V[(size_t)(j0 + r) * CH + hoff + c];
#pragma unroll
      for (int j = 0; j < 8; ++j) Vtt[(c + j) * 32 + r] = v[j];
    }
    wait_async_stage();
    __syncthreads();

    // ---- scores: 2 key tiles of 16, contraction over d=64 (2 WMMA steps) --
#pragma unroll
    for (int t = 0; t < 2; ++t) {
      v8f s = {};
#pragma unroll
      for (int kk = 0; kk < DH; kk += 32) {
        v16bf a  = load_a_frag(Aq + wm * DH + kk, DH);
        v16bf b  = load_b_kcontig(Kt + (t * 16) * DH + kk, DH);  // B[k][n]=Kt[n][k]
        s = WMMA_BF16(a, b, s);
        v16bf a2 = load_a_frag(Ae + wm * DH + kk, DH);
        v16bf b2 = load_b_kcontig(Qk + (t * 16) * DH + kk, DH);  // skew term
        s = WMMA_BF16(a2, b2, s);
      }
#pragma unroll
      for (int r = 0; r < 8; ++r)
        Sf[(wm + rbase + r) * 32 + t * 16 + col] = s[r] * scale;
    }
    __syncthreads();

    // ---- online softmax: one thread per query row ----
    if (tid < 64) {
      const int row = tid;
      float mo = mrow[row], mx = mo;
      for (int j = 0; j < 32; ++j) mx = fmaxf(mx, Sf[row * 32 + j]);
      float al = __expf(mo - mx), sum = 0.f;
      for (int j = 0; j < 32; ++j) {
        float p = __expf(Sf[row * 32 + j] - mx);
        Pt[row * 32 + j] = (bf16)p;
        sum += p;
      }
      lrow[row] = lrow[row] * al + sum;
      mrow[row] = mx;
      arow[row] = al;
    }
    __syncthreads();

    // ---- rescale accumulators and O += P.V ----
    v16bf a = load_a_frag(Pt + wm * 32, 32);
#pragma unroll
    for (int r = 0; r < 8; ++r) {
      float fa = arow[wm + rbase + r];
#pragma unroll
      for (int dt = 0; dt < 4; ++dt) acc[dt][r] *= fa;
    }
#pragma unroll
    for (int dt = 0; dt < 4; ++dt) {
      v16bf b = load_b_kcontig(Vtt + (dt * 16) * 32, 32);       // B[k][n]=Vtt[n][k]
      acc[dt] = WMMA_BF16(a, b, acc[dt]);
    }
  }
  __syncthreads();

#pragma unroll
  for (int dt = 0; dt < 4; ++dt)
#pragma unroll
    for (int r = 0; r < 8; ++r)
      O[(size_t)(i0 + wm + rbase + r) * CH + hoff + dt * 16 + col] =
          acc[dt][r] / lrow[wm + rbase + r];
}

// ---------------------------------------------------------------------------
// Host launcher
// ---------------------------------------------------------------------------
extern "C" void kernel_launch(void* const* d_in, const int* in_sizes, int n_in,
                              void* d_out, int out_size, void* d_ws, size_t ws_size,
                              hipStream_t stream) {
  (void)in_sizes; (void)n_in; (void)out_size; (void)ws_size;
  const float* x      = (const float*)d_in[0];   // [SEQ][CH]
  const float* wq     = (const float*)d_in[1];   // [CH][CH]
  const float* wk     = (const float*)d_in[2];
  const float* wv     = (const float*)d_in[3];
  const float* wo     = (const float*)d_in[4];
  const float* conv_w = (const float*)d_in[5];   // [CH][1][3]
  const float* er     = (const float*)d_in[6];   // [DH][SEQ]
  float* out = (float*)d_out;                    // [SEQ][CH]

  char* ws = (char*)d_ws;
  size_t off = 0;
  auto carve = [&](size_t bytes) -> void* {
    void* p = ws + off;
    off = (off + bytes + 255) & ~(size_t)255;
    return p;
  };

  bf16* Xb   = (bf16*)carve((size_t)SEQ * CH * 2);
  bf16* Wqb  = (bf16*)carve((size_t)CH * CH * 2);
  bf16* Wkb  = (bf16*)carve((size_t)CH * CH * 2);
  bf16* Wvb  = (bf16*)carve((size_t)CH * CH * 2);
  bf16* Wob  = (bf16*)carve((size_t)CH * CH * 2);
  bf16* ERtb = (bf16*)carve((size_t)SEQ * DH * 2);
  float* Qf  = (float*)carve((size_t)SEQ * CH * 4);
  float* Kf  = (float*)carve((size_t)SEQ * CH * 4);
  float* Vf  = (float*)carve((size_t)SEQ * CH * 4);
  bf16* Qcb  = (bf16*)carve((size_t)SEQ * CH * 2);
  bf16* Kcb  = (bf16*)carve((size_t)SEQ * CH * 2);
  bf16* Vcb  = (bf16*)carve((size_t)SEQ * CH * 2);
  float* Of  = (float*)carve((size_t)SEQ * CH * 4);
  bf16* Ob   = (bf16*)carve((size_t)SEQ * CH * 2);

  const int nXW = SEQ * CH;    // 2 Mi elements
  const int nWW = CH * CH;     // 1 Mi elements

  // 1) convert inputs to bf16
  cvt_f32_bf16<<<(nXW / 4 + 255) / 256, 256, 0, stream>>>(x,  Xb,  nXW / 4);
  cvt_f32_bf16<<<(nWW / 4 + 255) / 256, 256, 0, stream>>>(wq, Wqb, nWW / 4);
  cvt_f32_bf16<<<(nWW / 4 + 255) / 256, 256, 0, stream>>>(wk, Wkb, nWW / 4);
  cvt_f32_bf16<<<(nWW / 4 + 255) / 256, 256, 0, stream>>>(wv, Wvb, nWW / 4);
  cvt_f32_bf16<<<(nWW / 4 + 255) / 256, 256, 0, stream>>>(wo, Wob, nWW / 4);
  transpose_er<<<(SEQ * DH + 255) / 256, 256, 0, stream>>>(er, ERtb);

  // 2) QKV projections: [SEQ,CH] x [CH,CH]
  dim3 ggrid(CH / 64, SEQ / 64);
  gemm_bf16_nn<<<ggrid, 128, 0, stream>>>(Xb, Wqb, Qf, SEQ, CH, CH);
  gemm_bf16_nn<<<ggrid, 128, 0, stream>>>(Xb, Wkb, Kf, SEQ, CH, CH);
  gemm_bf16_nn<<<ggrid, 128, 0, stream>>>(Xb, Wvb, Vf, SEQ, CH, CH);

  // 3) depthwise conv (reference applies q_conv to q, k AND v)
  dwconv3_bf16<<<(nXW / 4 + 255) / 256, 256, 0, stream>>>(Qf, conv_w, Qcb);
  dwconv3_bf16<<<(nXW / 4 + 255) / 256, 256, 0, stream>>>(Kf, conv_w, Kcb);
  dwconv3_bf16<<<(nXW / 4 + 255) / 256, 256, 0, stream>>>(Vf, conv_w, Vcb);

  // 4) attention with relative-position skew
  attn_flash<<<dim3(SEQ / 64, HEADS), 128, 0, stream>>>(Qcb, Kcb, Vcb, ERtb, Of);

  // 5) output projection
  cvt_f32_bf16<<<(nXW / 4 + 255) / 256, 256, 0, stream>>>(Of, Ob, nXW / 4);
  gemm_bf16_nn<<<ggrid, 128, 0, stream>>>(Ob, Wob, out, SEQ, CH, CH);
}
